// _SelfAttentiveLBLBiLm_39367670235449
// MI455X (gfx1250) — matrix operand order repaired and checked by reading, not compiled, not run on hardware
//
#include <hip/hip_runtime.h>
#include <hip/hip_bf16.h>

// ---------------------------------------------------------------------------
// SelfAttentiveLBLBiLm forward for MI455X (gfx1250, wave32, WMMA).
//   L=2, H=8, D=512, S=1024, B=2, DK=64, WMAX=17 (band of 18 offsets), NHW=1
// All GEMMs run on v_wmma_f32_16x16x32_bf16 with a bf16 hi/lo split (3 WMMAs
// per tile) for ~fp32 accuracy. Attention exploits the 18-wide band: one
// wave32 per (b,h,i) row does scores+softmax+PV with ds_bpermute shuffles.
// ---------------------------------------------------------------------------

typedef __attribute__((ext_vector_type(16))) __bf16 v16bf;
typedef __attribute__((ext_vector_type(8)))  float  v8f;

// float -> bf16 (round to nearest even), and bf16 -> float
__device__ __forceinline__ unsigned short f2bf(float f) {
    unsigned u = __float_as_uint(f);
    unsigned r = u + 0x7FFFu + ((u >> 16) & 1u);
    return (unsigned short)(r >> 16);
}
__device__ __forceinline__ float bf2f(unsigned short h) {
    return __uint_as_float(((unsigned)h) << 16);
}

__device__ __forceinline__ v8f wmma_bf16(v16bf a, v16bf b, v8f c) {
    // (neg_a, A, neg_b, B, c_mod, C, reuse_a, reuse_b)
    return __builtin_amdgcn_wmma_f32_16x16x32_bf16(false, a, false, b, (short)0, c,
                                                   false, false);
}

// wave32 shuffles via ds_bpermute (avoids any wavefront-size macro issues)
__device__ __forceinline__ float shfl_f(float x, int srcLane) {
    return __int_as_float(__builtin_amdgcn_ds_bpermute(srcLane << 2, __float_as_int(x)));
}

// ---------------------------------------------------------------------------
// GEMM: Y[M,N] = X[M,K] @ W[K,N] + bias[N], fp32 in/out, bf16x2 WMMA inside.
// BM=BN=64, Kc=32, 128 threads (4 waves), each wave computes a 32x32 tile
// (2x2 WMMA 16x16 tiles). LDS staging writes fragments in a contiguous per-lane
// k order (k = 16*(lane/16) + element), shared by A and B:
//   A(16x32): lane<16 row m=lane holds K 0..15; lane>=16 holds K 16..31
//   B(32x16): lane<16 col n=lane holds K 0..15; lane>=16 holds K 16..31
// ---------------------------------------------------------------------------
__global__ void __launch_bounds__(128)
gemm_bias_bf16x2(const float* __restrict__ X, const float* __restrict__ W,
                 const float* __restrict__ bias, float* __restrict__ Y,
                 int M, int N, int K)
{
    __shared__ __attribute__((aligned(32))) unsigned short Ah[4][32][16];
    __shared__ __attribute__((aligned(32))) unsigned short Al[4][32][16];
    __shared__ __attribute__((aligned(32))) unsigned short Bh[4][32][16];
    __shared__ __attribute__((aligned(32))) unsigned short Bl[4][32][16];

    const int tid  = threadIdx.x;
    const int lane = tid & 31;
    const int wave = tid >> 5;
    const int wm   = wave >> 1;     // 0..1 : which 32-row half
    const int wn   = wave & 1;      // 0..1 : which 32-col half
    const int m0   = blockIdx.y * 64;
    const int n0   = blockIdx.x * 64;

    v8f acc00 = {}; v8f acc01 = {}; v8f acc10 = {}; v8f acc11 = {};

    for (int k0 = 0; k0 < K; k0 += 32) {
        __syncthreads();
        // ---- stage A block (64 rows x 32 k) with hi/lo bf16 split ----
        for (int idx = tid; idx < 64 * 32; idx += 128) {
            int m  = idx >> 5;
            int kk = idx & 31;
            float f = X[(size_t)(m0 + m) * K + k0 + kk];
            unsigned short hi = f2bf(f);
            unsigned short lo = f2bf(f - bf2f(hi));
            int grp = kk >> 3;                        // 0..3
            int ln  = (m & 15) | (kk & 16);           // +16 for K 16..31
            int ps  = kk & 15;
            Ah[m >> 4][ln][ps] = hi;
            Al[m >> 4][ln][ps] = lo;
        }
        // ---- stage B block (32 k x 64 cols) ----
        for (int idx = tid; idx < 32 * 64; idx += 128) {
            int n  = idx & 63;
            int kk = idx >> 6;
            float f = W[(size_t)(k0 + kk) * N + n0 + n];
            unsigned short hi = f2bf(f);
            unsigned short lo = f2bf(f - bf2f(hi));
            int ln = (n & 15) | (kk & 16);            // +16 for K 16..31
            int ps = kk & 15;
            Bh[n >> 4][ln][ps] = hi;
            Bl[n >> 4][ln][ps] = lo;
        }
        __syncthreads();

        v16bf a0h = *(const v16bf*)Ah[wm * 2 + 0][lane];
        v16bf a0l = *(const v16bf*)Al[wm * 2 + 0][lane];
        v16bf a1h = *(const v16bf*)Ah[wm * 2 + 1][lane];
        v16bf a1l = *(const v16bf*)Al[wm * 2 + 1][lane];
        v16bf b0h = *(const v16bf*)Bh[wn * 2 + 0][lane];
        v16bf b0l = *(const v16bf*)Bl[wn * 2 + 0][lane];
        v16bf b1h = *(const v16bf*)Bh[wn * 2 + 1][lane];
        v16bf b1l = *(const v16bf*)Bl[wn * 2 + 1][lane];

        // bf16x2: (hi+lo)*(hi+lo) ~= hi*hi + hi*lo + lo*hi
        acc00 = wmma_bf16(a0l, b0h, acc00);
        acc00 = wmma_bf16(a0h, b0l, acc00);
        acc00 = wmma_bf16(a0h, b0h, acc00);

        acc01 = wmma_bf16(a0l, b1h, acc01);
        acc01 = wmma_bf16(a0h, b1l, acc01);
        acc01 = wmma_bf16(a0h, b1h, acc01);

        acc10 = wmma_bf16(a1l, b0h, acc10);
        acc10 = wmma_bf16(a1h, b0l, acc10);
        acc10 = wmma_bf16(a1h, b0h, acc10);

        acc11 = wmma_bf16(a1l, b1h, acc11);
        acc11 = wmma_bf16(a1h, b1l, acc11);
        acc11 = wmma_bf16(a1h, b1h, acc11);
    }

    // ---- epilogue: C/D layout: lane n=lane&15; VGPR r -> row r (+8 lanes>=16)
    const int col0 = n0 + (wn * 2) * 16 + (lane & 15);
    const int col1 = col0 + 16;
    const int row0 = m0 + (wm * 2) * 16 + ((lane >> 4) << 3);
    const int row1 = row0 + 16;
    const float bc0 = bias[col0];
    const float bc1 = bias[col1];
#pragma unroll
    for (int r = 0; r < 8; ++r) {
        Y[(size_t)(row0 + r) * N + col0] = acc00[r] + bc0;
        Y[(size_t)(row0 + r) * N + col1] = acc01[r] + bc1;
        Y[(size_t)(row1 + r) * N + col0] = acc10[r] + bc0;
        Y[(size_t)(row1 + r) * N + col1] = acc11[r] + bc1;
    }
}

// ---------------------------------------------------------------------------
// Banded attention: one wave32 per (b,h,i). Lanes 0..17 own one band offset
// each (score + bias), wave shuffles do softmax, then all 32 lanes compute
// the 64-dim weighted sum of V (2 dims per lane).
// ---------------------------------------------------------------------------
__global__ void __launch_bounds__(256)
band_attn(const float* __restrict__ q, const float* __restrict__ k,
          const float* __restrict__ v, const float* __restrict__ rel,
          float* __restrict__ out, int backward)
{
    const int tid  = threadIdx.x;
    const int lane = tid & 31;
    const int g    = blockIdx.x * 8 + (tid >> 5);   // 0 .. B*H*S-1
    const int i    = g & 1023;
    const int h    = (g >> 10) & 7;
    const int b    = g >> 13;
    const int row  = (b << 10) + i;
    const float* qp = q + (size_t)row * 512 + h * 64;

    // scores for band offsets jj = lane (0..17); col index into rel == jj
    float s = -1e9f;
    {
        int j = backward ? (i + lane) : (i - 17 + lane);
        bool valid = (lane < 18) && (j >= 0) && (j < 1024);
        if (valid) {
            const float4* q4 = (const float4*)qp;
            const float4* k4 = (const float4*)(k + (size_t)((b << 10) + j) * 512 + h * 64);
            float acc = 0.f;
#pragma unroll
            for (int d = 0; d < 16; ++d) {
                float4 a = q4[d], c = k4[d];
                acc += a.x * c.x + a.y * c.y + a.z * c.z + a.w * c.w;
            }
            s = acc * 0.125f + rel[h * 18 + lane];   // 1/sqrt(64) = 0.125
        }
    }
    // wave-wide softmax (invalid lanes carry -1e9 -> exp underflows to 0)
    float m = s;
#pragma unroll
    for (int off = 16; off > 0; off >>= 1) m = fmaxf(m, shfl_f(m, lane ^ off));
    float e = __expf(s - m);
    float sum = e;
#pragma unroll
    for (int off = 16; off > 0; off >>= 1) sum += shfl_f(sum, lane ^ off);
    float p = e / sum;

    // out[d] = sum_j p_j * v[j][d]; lane handles d=lane and d=lane+32
    float o0 = 0.f, o1 = 0.f;
    for (int jj = 0; jj < 18; ++jj) {
        float pj = shfl_f(p, jj);
        int j = backward ? (i + jj) : (i - 17 + jj);
        if (j >= 0 && j < 1024) {
            const float* vp = v + (size_t)((b << 10) + j) * 512 + h * 64;
            o0 += pj * vp[lane];
            o1 += pj * vp[lane + 32];
        }
    }
    float* op = out + (size_t)row * 512 + h * 64;
    op[lane]      = o0;
    op[lane + 32] = o1;
}

// ---------------------------------------------------------------------------
// Highway combine + residual + write output slice.
//   hw[row, 0:512] = nonlinear, hw[row, 512:1024] = gate logits
//   val = sigmoid(gate)*proj + (1-sigmoid(gate))*relu(nl)  (+ cache if l>0)
//   x (in/out) holds layer input (cache) on entry, next-layer input on exit.
// ---------------------------------------------------------------------------
__global__ void __launch_bounds__(256)
highway_combine(const float* __restrict__ proj, const float* __restrict__ hw,
                float* __restrict__ x, float* __restrict__ outSlice, int addCache)
{
    const size_t idx = (size_t)blockIdx.x * 256 + threadIdx.x;   // 0 .. 2048*512-1
    const int row = (int)(idx >> 9);
    const int c   = (int)(idx & 511);
    float nl = hw[((size_t)row << 10) + c];
    float gz = hw[((size_t)row << 10) + 512 + c];
    float g  = 1.f / (1.f + __expf(-gz));
    float val = g * proj[idx] + (1.f - g) * fmaxf(nl, 0.f);
    if (addCache) val += x[idx];
    x[idx] = val;
    outSlice[((size_t)row << 10) + c] = val;   // output stride 2D=1024
}

__global__ void __launch_bounds__(256)
dup_copy(const float* __restrict__ src, float* __restrict__ a, float* __restrict__ b)
{
    const size_t i = (size_t)blockIdx.x * 256 + threadIdx.x;
    float f = src[i];
    a[i] = f;
    b[i] = f;
}

// ---------------------------------------------------------------------------
// Input order (setup_inputs): 0 inputs, 1 masks,
//   fwd: 2 Wq, 3 bq, 4 Wk, 5 bk, 6 Wv, 7 bv, 8 Wo, 9 bo, 10 rel, 11 hw_W, 12 hw_b
//   bwd: 13..23 same layout.
// Output: [L, B, S, 2D] fp32.
// ---------------------------------------------------------------------------
extern "C" void kernel_launch(void* const* d_in, const int* in_sizes, int n_in,
                              void* d_out, int out_size, void* d_ws, size_t ws_size,
                              hipStream_t stream)
{
    (void)in_sizes; (void)n_in; (void)out_size; (void)ws_size;
    const int L = 2, H = 8, D = 512, S = 1024, B = 2;
    const int M = B * S;                       // 2048 token rows
    const size_t T = (size_t)M * D;            // 1M floats per activation

    float* ws   = (float*)d_ws;
    float* xf   = ws;            // fwd stream activations
    float* xb   = ws + 1 * T;    // bwd stream activations
    float* qb   = ws + 2 * T;
    float* kb   = ws + 3 * T;
    float* vb   = ws + 4 * T;
    float* att  = ws + 5 * T;
    float* proj = ws + 6 * T;
    float* hw   = ws + 7 * T;    // 2T (N = 2D)
    float* out  = (float*)d_out;
    const float* inp = (const float*)d_in[0];

    dup_copy<<<dim3((unsigned)(T / 256)), 256, 0, stream>>>(inp, xf, xb);

    const dim3 g512(D / 64, M / 64);           // (8, 32)
    const dim3 g1024(2 * D / 64, M / 64);      // (16, 32)

    for (int l = 0; l < L; ++l) {
        for (int dir = 0; dir < 2; ++dir) {
            const int base = 2 + dir * 11;
            const float* Wq  = (const float*)d_in[base + 0] + (size_t)l * D * D;
            const float* bq  = (const float*)d_in[base + 1] + (size_t)l * D;
            const float* Wk  = (const float*)d_in[base + 2] + (size_t)l * D * D;
            const float* bk  = (const float*)d_in[base + 3] + (size_t)l * D;
            const float* Wv  = (const float*)d_in[base + 4] + (size_t)l * D * D;
            const float* bv  = (const float*)d_in[base + 5] + (size_t)l * D;
            const float* Wo  = (const float*)d_in[base + 6] + (size_t)l * D * D;
            const float* bo  = (const float*)d_in[base + 7] + (size_t)l * D;
            const float* rel = (const float*)d_in[base + 8] + (size_t)l * H * 18;
            const float* hwW = (const float*)d_in[base + 9] + (size_t)l * D * 2 * D;
            const float* hwb = (const float*)d_in[base + 10] + (size_t)l * 2 * D;
            float* x = dir ? xb : xf;

            gemm_bias_bf16x2<<<g512, 128, 0, stream>>>(x, Wq, bq, qb, M, D, D);
            gemm_bias_bf16x2<<<g512, 128, 0, stream>>>(x, Wk, bk, kb, M, D, D);
            gemm_bias_bf16x2<<<g512, 128, 0, stream>>>(x, Wv, bv, vb, M, D, D);
            band_attn<<<dim3(M * H / 8), 256, 0, stream>>>(qb, kb, vb, rel, att, dir);
            gemm_bias_bf16x2<<<g512, 128, 0, stream>>>(att, Wo, bo, proj, M, D, D);
            gemm_bias_bf16x2<<<g1024, 128, 0, stream>>>(proj, hwW, hwb, hw, M, 2 * D, D);
            highway_combine<<<dim3((unsigned)(T / 256)), 256, 0, stream>>>(
                proj, hw, x, out + (size_t)l * M * 2 * D + (size_t)dir * D, l != 0);
        }
    }
}
